// MinimalGRU_69269232550448
// MI455X (gfx1250) — compile-verified, matrix operand order checked by reading, not run on hardware
//
#include <hip/hip_runtime.h>
#include <stdint.h>

// ---------------- problem constants ----------------
static constexpr int BATCH = 32;
static constexpr int SEQ   = 512;
static constexpr int HDIM  = 1024;
static constexpr int D0    = 512;            // layer-0 input width
static constexpr int GATES = 2 * HDIM;       // 2048 gate rows

// ---------------- persistent-kernel geometry ----------------
static constexpr int BLK_PER_LAYER = 128;    // each block owns 8 h-columns
static constexpr int NBLK    = 2 * BLK_PER_LAYER; // both layers pipelined
static constexpr int THREADS = 64;           // 2 waves (wave32): one per M-tile of 16
static constexpr int STAGGER = 2;            // layer-1 runs 2 steps behind layer-0

// ---------------- types ----------------
typedef __bf16 bf16_t;
typedef __attribute__((ext_vector_type(8)))  __bf16 v8bf;
typedef __attribute__((ext_vector_type(16))) __bf16 v16bf;
typedef __attribute__((ext_vector_type(8)))  float  v8f;

union Frag16 { v16bf v; v8bf h[2]; };

__device__ __forceinline__ unsigned short f2bf_rne(float f) {
  unsigned u = __float_as_uint(f);
  u += 0x7FFFu + ((u >> 16) & 1u);     // round to nearest even
  return (unsigned short)(u >> 16);
}

// ---------------- prep kernels ----------------
__global__ void k_f2bf(const float* __restrict__ src, unsigned short* __restrict__ dst, int n) {
  int i = blockIdx.x * blockDim.x + threadIdx.x;
  if (i < n) dst[i] = f2bf_rne(src[i]);
}

// Wcat[row, :] = [ w_ih[row, 0:K1] | w_hh[row, 0:H] ]  as bf16, rows = 2048
__global__ void k_build_wcat(const float* __restrict__ wih, const float* __restrict__ whh,
                             unsigned short* __restrict__ dst, int K1) {
  const int K = K1 + HDIM;
  long long i = (long long)blockIdx.x * blockDim.x + threadIdx.x;
  long long n = (long long)GATES * K;
  if (i >= n) return;
  int row = (int)(i / K);
  int k   = (int)(i - (long long)row * K);
  float v = (k < K1) ? wih[(long long)row * K1 + k] : whh[(long long)row * HDIM + (k - K1)];
  dst[i] = f2bf_rne(v);
}

__global__ void k_bias(const float* __restrict__ a, const float* __restrict__ c,
                       float* __restrict__ d, int n) {
  int i = blockIdx.x * blockDim.x + threadIdx.x;
  if (i < n) d[i] = a[i] + c[i];
}

__global__ void k_init_barrier(unsigned* bar) {
  if (threadIdx.x < 2) bar[threadIdx.x * 64] = 0u;   // cnt @ bar[0], gen @ bar[64]
}

// ---------------- grid-wide barrier (persistent kernel) ----------------
__device__ __forceinline__ void grid_barrier(unsigned* bar, unsigned nb) {
  __syncthreads();
  if (threadIdx.x == 0) {
    __threadfence();                               // release our stores to device scope
    unsigned* cnt = bar;
    unsigned* gen = bar + 64;
    unsigned g = atomicAdd(gen, 0u);
    if (atomicAdd(cnt, 1u) == nb - 1u) {
      atomicExch(cnt, 0u);
      __threadfence();
      atomicAdd(gen, 1u);
    } else {
      while (atomicAdd(gen, 0u) == g) { __builtin_amdgcn_s_sleep(2); }
    }
  }
  __syncthreads();
  __threadfence();                                 // acquire: don't read stale h/y lines
}

// ---------------- the pipelined recurrent scan ----------------
// Grid: 256 blocks. blocks [0,128)=layer0 (t = s), [128,256)=layer1 (t = s-2).
// Block `blk` of a layer owns h columns [8*blk, 8*blk+8): gate rows
//   ug: 8*blk..8*blk+7   and   og: 1024+8*blk..1024+8*blk+7
// Those 16 gate rows' Wcat weights (16 x K bf16, 64KB max) live in LDS for the whole
// scan, xor-swizzled so B-fragment ds_load_b128s hit the 2-way-conflict floor.
//
// Critical-path split: the input projection (bias + x_t . W_ih^T) does not depend on
// the barrier-published h, so it is computed into a *pending* accumulator at the END
// of the previous step (overlapping the barrier wait). Only the recurrent K=1024 GEMM
// sits between barriers. Layer 1 therefore trails layer 0 by 2 steps so that its
// pending x-part for t reads y0bf[t] one full step after layer 0 published it.
__launch_bounds__(THREADS)
__global__ void k_gru_scan(
    const float* __restrict__ hx,              // [2,32,1024] fp32
    const unsigned short* __restrict__ xbf,    // [32,512,512]  bf16
    const unsigned short* __restrict__ wcat0,  // [2048,1536]   bf16
    const float* __restrict__ bias0,           // [2048]
    const unsigned short* __restrict__ wcat1,  // [2048,2048]   bf16
    const float* __restrict__ bias1,           // [2048]
    unsigned short* __restrict__ y0bf,         // [32,512,1024] bf16 (layer0 out / layer1 in)
    float* __restrict__ out,                   // [32,512,1024] fp32 (final output)
    unsigned short* __restrict__ hbuf,         // [2 layers][2 parity][32][1024] bf16
    unsigned* __restrict__ bar)
{
  __shared__ unsigned char lds[16 * 4096];     // 64KB: 16 weight rows, 4KB row stride

  const int tid  = threadIdx.x;
  const int lane = tid & 31;                   // wave32 (gfx1250)
  const int w    = tid >> 5;                   // wave id = M-tile (A rows 16w..16w+15)
  const int lay  = blockIdx.x >> 7;            // layer 0/1
  const int blk  = blockIdx.x & (BLK_PER_LAYER - 1);

  const int K1 = lay ? HDIM : D0;
  const int K  = K1 + HDIM;
  const unsigned short* wcat = lay ? wcat1 : wcat0;
  const float*          bias = lay ? bias1 : bias0;
  const unsigned short* asrc = lay ? y0bf  : xbf;   // activation source rows [m, t, 0:K1]

  // ---- stage this block's 16 Wcat rows into LDS (xor-swizzle bits[7:4] by row) ----
  {
    const int chunksPerRow = K / 8;            // 16B chunks
    for (int idx = tid; idx < 16 * chunksPerRow; idx += THREADS) {
      int row   = idx / chunksPerRow;
      int kElem = (idx - row * chunksPerRow) * 8;
      int grow  = (row < 8) ? (8 * blk + row) : (HDIM + 8 * blk + (row - 8));
      v8bf vv = *(const v8bf*)(wcat + (size_t)grow * K + kElem);
      unsigned off = (unsigned)row * 4096u +
                     (((unsigned)kElem * 2u) ^ (((unsigned)row) << 4));
      *(v8bf*)(lds + off) = vv;
    }
  }
  __syncthreads();                             // staging must be visible to BOTH waves
                                               // before the prologue cPend GEMM below

  // ---- per-lane constants ----
  const int  n16   = lane & 15;                // tile column / B-fragment weight row
  const int  hi    = lane >> 4;                // K-split half (0/1)
  const int  koff  = hi << 3;                  // +8 elements for lanes 16..31
  const int  growN = (n16 < 8) ? (8 * blk + n16) : (HDIM + 8 * blk + (n16 - 8));
  const float cbias = bias[growN];             // hoisted: C init value, constant per lane
  const int  aRowM = (w << 4) + n16;           // A-fragment row this lane loads
  const int  jcol  = 8 * blk + (lane & 7);     // h column this lane updates (if active)
  const bool activeLane = (lane & 8) == 0;
  const unsigned ldsRowBase = (unsigned)n16 * 4096u;
  const unsigned ldsSw      = ((unsigned)n16) << 4;

  // ---- h lives in registers: 8 rows (m = 16w + r + 8*hi) of this lane's column ----
  float hreg[8];
  {
    const float* hx_l = hx + (size_t)lay * BATCH * HDIM;
    unsigned short* hb0 = hbuf + (size_t)(lay * 2 + 0) * (BATCH * HDIM);
    #pragma unroll
    for (int r = 0; r < 8; ++r) {
      int m = (w << 4) + r + (hi << 3);
      hreg[r] = activeLane ? hx_l[m * HDIM + jcol] : 0.f;
      if (activeLane) hb0[m * HDIM + jcol] = f2bf_rne(hreg[r]);
    }
  }

  const size_t aBaseM = (size_t)aRowM * SEQ * K1;   // row m, t=0 in asrc

  // pending accumulator: bias + x_t . W_ih^T for the NEXT consumed timestep
  v8f cPend;
  #pragma unroll
  for (int i = 0; i < 8; ++i) cPend[i] = cbias;
  if (lay == 0) {                                   // layer0 consumes t=0 at s=0
    const unsigned short* aRow = asrc + aBaseM;     // t = 0
    #pragma unroll 4
    for (int kk = 0; kk < K1; kk += 32) {
      Frag16 a, bb;
      a.h[0] = *(const v8bf*)(aRow + kk + koff);
      a.h[1] = *(const v8bf*)(aRow + kk + 16 + koff);
      bb.h[0] = *(const v8bf*)(lds + ldsRowBase + ((((unsigned)(kk + koff)) * 2u) ^ ldsSw));
      bb.h[1] = *(const v8bf*)(lds + ldsRowBase + ((((unsigned)(kk + 16 + koff)) * 2u) ^ ldsSw));
      cPend = __builtin_amdgcn_wmma_f32_16x16x32_bf16(false, a.v, false, bb.v,
                                                      (short)0, cPend, false, false);
    }
  }
  grid_barrier(bar, NBLK);                     // publish initial h (and wait for everyone)

  for (int s = 0; s < SEQ + STAGGER; ++s) {
    // ---------- consume: recurrent half + activation for t ----------
    const int tc = (lay == 0) ? s : s - STAGGER;
    if (tc >= 0 && tc < SEQ) {
      v8f c = cPend;
      const unsigned short* hRow =
          hbuf + (size_t)(lay * 2 + (tc & 1)) * (BATCH * HDIM) + (size_t)aRowM * HDIM;
      #pragma unroll 4
      for (int kk = 0; kk < HDIM; kk += 32) {
        Frag16 a, bb;
        a.h[0] = *(const v8bf*)(hRow + kk + koff);
        a.h[1] = *(const v8bf*)(hRow + kk + 16 + koff);
        const unsigned kl = (unsigned)(K1 + kk);
        bb.h[0] = *(const v8bf*)(lds + ldsRowBase + (((kl + koff) * 2u) ^ ldsSw));
        bb.h[1] = *(const v8bf*)(lds + ldsRowBase + (((kl + 16u + koff) * 2u) ^ ldsSw));
        c = __builtin_amdgcn_wmma_f32_16x16x32_bf16(false, a.v, false, bb.v,
                                                    (short)0, c, false, false);
      }

      // gate combine: lane n holds ug, lane n+8 holds og for the same h column
      float newh[8];
      #pragma unroll
      for (int r = 0; r < 8; ++r) {
        float mine  = c[r];
        float other = __shfl_xor(mine, 8, 32);          // uniform: all lanes execute
        float ug = activeLane ? mine  : other;
        float og = activeLane ? other : mine;
        ug = 1.f / (1.f + __expf(-ug));                 // sigmoid
        og = og > 0.f ? og : 0.f;                       // relu
        newh[r] = ug * hreg[r] + (1.f - ug) * og;
        hreg[r] = newh[r];
      }

      unsigned short* hbn =
          hbuf + (size_t)(lay * 2 + ((tc + 1) & 1)) * (BATCH * HDIM);
      if (activeLane) {
        #pragma unroll
        for (int r = 0; r < 8; ++r) {
          int m = (w << 4) + r + (hi << 3);
          unsigned short hb = f2bf_rne(newh[r]);
          hbn[m * HDIM + jcol] = hb;                    // h for next step (all blocks read)
          size_t o = ((size_t)m * SEQ + tc) * HDIM + jcol;
          if (lay == 0) y0bf[o] = hb;                   // feeds layer 1 (read at s=tc+1)
          else          __builtin_nontemporal_store(newh[r], &out[o]); // streamed once
        }
      }
    }

    // ---------- produce: pending x-part for the next consumed timestep ----------
    // layer0: t_next = s+1 (xbf always ready). layer1: t_next = s-1 (y0bf[s-1] was
    // published at the end of step s-1, i.e. before this step began).
    const int tn = (lay == 0) ? s + 1 : s - 1;
    if (tn >= 0 && tn < SEQ) {
      v8f cp;
      #pragma unroll
      for (int i = 0; i < 8; ++i) cp[i] = cbias;
      const unsigned short* aRow = asrc + aBaseM + (size_t)tn * K1;
      #pragma unroll 4
      for (int kk = 0; kk < K1; kk += 32) {
        Frag16 a, bb;
        a.h[0] = *(const v8bf*)(aRow + kk + koff);
        a.h[1] = *(const v8bf*)(aRow + kk + 16 + koff);
        bb.h[0] = *(const v8bf*)(lds + ldsRowBase + ((((unsigned)(kk + koff)) * 2u) ^ ldsSw));
        bb.h[1] = *(const v8bf*)(lds + ldsRowBase + ((((unsigned)(kk + 16 + koff)) * 2u) ^ ldsSw));
        cp = __builtin_amdgcn_wmma_f32_16x16x32_bf16(false, a.v, false, bb.v,
                                                     (short)0, cp, false, false);
      }
      cPend = cp;
    }

    grid_barrier(bar, NBLK);
  }
}

// ---------------- host side ----------------
extern "C" void kernel_launch(void* const* d_in, const int* in_sizes, int n_in,
                              void* d_out, int out_size, void* d_ws, size_t ws_size,
                              hipStream_t stream) {
  (void)in_sizes; (void)n_in; (void)out_size; (void)ws_size;
  const float* x    = (const float*)d_in[0];
  const float* hx   = (const float*)d_in[1];
  const float* wih0 = (const float*)d_in[2];
  const float* whh0 = (const float*)d_in[3];
  const float* bih0 = (const float*)d_in[4];
  const float* bhh0 = (const float*)d_in[5];
  const float* wih1 = (const float*)d_in[6];
  const float* whh1 = (const float*)d_in[7];
  const float* bih1 = (const float*)d_in[8];
  const float* bhh1 = (const float*)d_in[9];
  float* out = (float*)d_out;

  // workspace carve (256B aligned)
  char* wp = (char*)d_ws;
  auto carve = [&](size_t bytes) -> char* {
    char* p = wp; wp += (bytes + 255) & ~(size_t)255; return p;
  };
  unsigned short* xbf  = (unsigned short*)carve((size_t)BATCH * SEQ * D0 * 2);    // 16 MB
  unsigned short* y0bf = (unsigned short*)carve((size_t)BATCH * SEQ * HDIM * 2);  // 32 MB
  unsigned short* wc0  = (unsigned short*)carve((size_t)GATES * (D0 + HDIM) * 2); //  6 MB
  unsigned short* wc1  = (unsigned short*)carve((size_t)GATES * (2 * HDIM) * 2);  //  8 MB
  float*          bs0  = (float*)carve(GATES * sizeof(float));
  float*          bs1  = (float*)carve(GATES * sizeof(float));
  unsigned short* hbuf = (unsigned short*)carve((size_t)2 * 2 * BATCH * HDIM * 2);
  unsigned*       bar  = (unsigned*)carve(512);

  // prep passes
  {
    int n = BATCH * SEQ * D0;
    k_f2bf<<<(n + 255) / 256, 256, 0, stream>>>(x, xbf, n);
  }
  {
    long long n = (long long)GATES * (D0 + HDIM);
    k_build_wcat<<<(int)((n + 255) / 256), 256, 0, stream>>>(wih0, whh0, wc0, D0);
  }
  {
    long long n = (long long)GATES * (2 * HDIM);
    k_build_wcat<<<(int)((n + 255) / 256), 256, 0, stream>>>(wih1, whh1, wc1, HDIM);
  }
  k_bias<<<(GATES + 255) / 256, 256, 0, stream>>>(bih0, bhh0, bs0, GATES);
  k_bias<<<(GATES + 255) / 256, 256, 0, stream>>>(bih1, bhh1, bs1, GATES);
  k_init_barrier<<<1, 64, 0, stream>>>(bar);

  // persistent pipelined scan: both layers, staggered by two timesteps
  k_gru_scan<<<NBLK, THREADS, 0, stream>>>(hx, xbf, wc0, bs0, wc1, bs1,
                                           y0bf, out, hbuf, bar);
}
